// TransformerHead_35794257445458
// MI455X (gfx1250) — matrix operand order repaired and checked by reading, not compiled, not run on hardware
//
#include <hip/hip_runtime.h>

#define Bsz 2
#define Sq  4096
#define Hn  8
#define DHd 64
#define Dm  512
#define Mr  (Bsz*Sq)   // 8192 rows

typedef unsigned short u16;
typedef __attribute__((ext_vector_type(16))) __bf16 v16bf;
typedef __attribute__((ext_vector_type(8)))  float  v8f;
typedef __attribute__((ext_vector_type(8)))  u16    v8u;
typedef __attribute__((ext_vector_type(8)))  unsigned v8ui;
typedef __attribute__((ext_vector_type(4)))  unsigned v4ui;
typedef __attribute__((ext_vector_type(4)))  float  v4f;

union FragBF { v16bf bf; v8u h[2]; v8ui d; u16 u[16]; };

// round-to-nearest-even f32 -> bf16 (used in one-time prep/projection epilogues)
static __device__ __forceinline__ u16 f2bf(float f) {
  unsigned u = __float_as_uint(f);
  u += 0x7FFFu + ((u >> 16) & 1u);
  return (u16)(u >> 16);
}

// pack two f32 into two truncated bf16 with a single v_perm_b32 (hot path)
static __device__ __forceinline__ unsigned pk_bf16_trunc(float lo, float hi) {
  return __builtin_amdgcn_perm(__float_as_uint(hi), __float_as_uint(lo), 0x07060302u);
}

static __device__ __forceinline__ float fexp2(float x) {
#if __has_builtin(__builtin_amdgcn_exp2f)
  return __builtin_amdgcn_exp2f(x);   // single v_exp_f32
#else
  return exp2f(x);
#endif
}

// A-operand fragment (16x32 bf16, row-major source, row stride ld elements).
// Lane half 0: K = kb+0..15 ; half 1: K = kb+16..31
static __device__ __forceinline__ v16bf load_fragA(const u16* base, int row, int ld,
                                                   int kb, int lane) {
  const int hf = lane >> 4;
  const u16* p = base + (size_t)row * ld + (kb + 16 * hf);
  FragBF f;
  f.h[0] = *(const v8u*)(p);
  f.h[1] = *(const v8u*)(p + 8);
  return f.bf;
}

// B-operand fragment (32x16 bf16). Source stored as [N][K] row-major ("transposed"),
// so each lane reads 32 contiguous bytes of its column's K-slice.
// Lane half 0: K = kb+0..15 ; half 1: K = kb+16..31
static __device__ __forceinline__ v16bf load_fragB(const u16* base, int ncol, int ld,
                                                   int kb, int lane) {
  const int hf = lane >> 4;
  const u16* p = base + (size_t)ncol * ld + (kb + 16 * hf);
  FragBF f;
  f.h[0] = *(const v8u*)(p);
  f.h[1] = *(const v8u*)(p + 8);
  return f.bf;
}

static __device__ __forceinline__ v8f wmma_bf16(v16bf a, v16bf b, v8f c) {
  return __builtin_amdgcn_wmma_f32_16x16x32_bf16(false, a, false, b, (short)0, c,
                                                 false, false);
}

// ---------------- prep kernels ----------------

__global__ void cvt_f32_to_bf16(const float* __restrict__ x, u16* __restrict__ y) {
  const int i = (blockIdx.x * blockDim.x + threadIdx.x) * 8;
  v4f a = *(const v4f*)(x + i);
  v4f b = *(const v4f*)(x + i + 4);
  v8u o;
  o[0] = f2bf(a[0]); o[1] = f2bf(a[1]); o[2] = f2bf(a[2]); o[3] = f2bf(a[3]);
  o[4] = f2bf(b[0]); o[5] = f2bf(b[1]); o[6] = f2bf(b[2]); o[7] = f2bf(b[3]);
  *(v8u*)(y + i) = o;
}

// W (f32, [K=512][N=512]) -> Wt (bf16, [N][K]) so B fragments are contiguous
__global__ void transpose_w_bf16(const float* __restrict__ w, u16* __restrict__ wt) {
  const int t = blockIdx.x * blockDim.x + threadIdx.x;  // 0..512*512-1
  const int n = t >> 9, k = t & 511;
  wt[t] = f2bf(w[k * Dm + n]);
}

// ---------------- fused GEMM (+bias, +scale, layout epilogue) ----------------
// C[M=8192, N=512] = A[M,512] @ Wt^T, A bf16 row-major, Wt bf16 [N][K].
// 8 waves, wave tile 32x64 (2x4 WMMA tiles), WG tile 128x128.
// mode 0: store bf16 as [(b*8+h)][seq][64]   (Q / K; Q pre-scaled by log2e/8)
// mode 1: store bf16 as [(b*8+h)][64][seq]   (V transposed)
// mode 2: store f32 row-major [M][512]       (final output)
#define MODE_QK 0
#define MODE_VT 1
#define MODE_F32 2

__global__ __launch_bounds__(256) void gemm_bf16(const u16* __restrict__ A,
                                                 const u16* __restrict__ Bt,
                                                 const float* __restrict__ bias,
                                                 void* __restrict__ out,
                                                 int mode, float scale) {
  const int lane = threadIdx.x & 31;
  const int wave = threadIdx.x >> 5;
  const int wr = wave >> 1, wc = wave & 1;
  const int lr = lane & 15, hf = lane >> 4;
  const int row0 = blockIdx.x * 128 + wr * 32;
  const int col0 = blockIdx.y * 128 + wc * 64;

  v8f acc[2][4] = {};
  for (int kb = 0; kb < Dm; kb += 32) {
    v16bf af[2], bfm[4];
#pragma unroll
    for (int i = 0; i < 2; ++i) af[i] = load_fragA(A, row0 + 16 * i + lr, Dm, kb, lane);
#pragma unroll
    for (int j = 0; j < 4; ++j) bfm[j] = load_fragB(Bt, col0 + 16 * j + lr, Dm, kb, lane);
#pragma unroll
    for (int i = 0; i < 2; ++i)
#pragma unroll
      for (int j = 0; j < 4; ++j) acc[i][j] = wmma_bf16(af[i], bfm[j], acc[i][j]);
  }

#pragma unroll
  for (int i = 0; i < 2; ++i) {
#pragma unroll
    for (int j = 0; j < 4; ++j) {
      const int col = col0 + 16 * j + lr;          // fixed per lane (D layout)
      const float bv = bias[col];
      const int rowb = row0 + 16 * i + 8 * hf;     // rows rowb..rowb+7 in acc[0..7]
      if (mode == MODE_VT) {
        const int bh = (rowb >> 12) * Hn + (col >> 6);
        const int d = col & 63;
        const int seq0 = rowb & (Sq - 1);
        v8u o;
#pragma unroll
        for (int r = 0; r < 8; ++r) o[r] = f2bf((acc[i][j][r] + bv) * scale);
        *(v8u*)((u16*)out + ((size_t)bh * DHd + d) * Sq + seq0) = o;
      } else if (mode == MODE_QK) {
        const int bh = (rowb >> 12) * Hn + (col >> 6);
        const int d = col & 63;
        u16* o = (u16*)out + ((size_t)bh * Sq + (rowb & (Sq - 1))) * DHd + d;
#pragma unroll
        for (int r = 0; r < 8; ++r) o[(size_t)r * DHd] = f2bf((acc[i][j][r] + bv) * scale);
      } else {
        float* o = (float*)out + (size_t)rowb * Dm + col;
#pragma unroll
        for (int r = 0; r < 8; ++r) o[(size_t)r * Dm] = (acc[i][j][r] + bv) * scale;
      }
    }
  }
}

// ---------------- flash attention (bf16 WMMA, exp2-domain online softmax) -------
// Per wave: 32 queries (2 query tiles). Computes S^T = K @ Q^T so the score D-tiles
// map (with one xor-16 half exchange of the packed dwords) directly onto the B
// operand of O^T = V^T @ P^T. Scores are in log2 domain (log2e folded into Q).
#define QPT 2

__global__ __launch_bounds__(128) void flash_attn(const u16* __restrict__ Qb,
                                                  const u16* __restrict__ Kb,
                                                  const u16* __restrict__ Vt,
                                                  u16* __restrict__ ctx) {
  const int lane = threadIdx.x & 31;
  const int wave = threadIdx.x >> 5;
  const int lr = lane & 15, hf = lane >> 4;
  const int bh = blockIdx.y;
  const int q0 = blockIdx.x * (4 * QPT * 16) + wave * (QPT * 16);

  const u16* Qh = Qb + (size_t)bh * Sq * DHd;   // [seq][64]  (pre-scaled)
  const u16* Kh = Kb + (size_t)bh * Sq * DHd;   // [seq][64]
  const u16* Vh = Vt + (size_t)bh * DHd * Sq;   // [64][seq]  (transposed)

  // Q^T as B operands: 2 d-chunks per query tile, resident for the whole K loop
  v16bf qf[QPT][2];
#pragma unroll
  for (int qt = 0; qt < QPT; ++qt)
#pragma unroll
    for (int c = 0; c < 2; ++c)
      qf[qt][c] = load_fragB(Qh, q0 + 16 * qt + lr, DHd, 32 * c, lane);

  v8f oacc[4][QPT];   // O^T tiles: dim-chunk x query-tile
#pragma unroll
  for (int dt = 0; dt < 4; ++dt)
#pragma unroll
    for (int qt = 0; qt < QPT; ++qt) oacc[dt][qt] = (v8f){};
  float mrow[QPT], lrow[QPT];
#pragma unroll
  for (int qt = 0; qt < QPT; ++qt) { mrow[qt] = -1e30f; lrow[qt] = 0.f; }

  for (int kb = 0; kb < Sq; kb += 32) {
    // prefetch next key block of K and V^T into the cache hierarchy
    if (kb + 32 < Sq) {
      __builtin_prefetch(Kh + (size_t)(kb + 32 + lane) * DHd, 0, 3);
      __builtin_prefetch(Vh + (size_t)lane * Sq + (kb + 32), 0, 3);
      __builtin_prefetch(Vh + (size_t)(lane + 32) * Sq + (kb + 32), 0, 3);
    }

    // S^T tiles: [16 keys x 16 queries]; kt picks keys kb+16*kt
    v8f st[2][QPT];
#pragma unroll
    for (int kt = 0; kt < 2; ++kt)
#pragma unroll
      for (int qt = 0; qt < QPT; ++qt) st[kt][qt] = (v8f){};
#pragma unroll
    for (int c = 0; c < 2; ++c) {
#pragma unroll
      for (int kt = 0; kt < 2; ++kt) {
        v16bf kf = load_fragA(Kh, kb + 16 * kt + lr, DHd, 32 * c, lane);
#pragma unroll
        for (int qt = 0; qt < QPT; ++qt)
          st[kt][qt] = wmma_bf16(kf, qf[qt][c], st[kt][qt]);
      }
    }

    v16bf pf[QPT];
#pragma unroll
    for (int qt = 0; qt < QPT; ++qt) {
      // per-query (per-lane-column) block max over 32 keys (log2-domain scores)
      float bm = -1e30f;
#pragma unroll
      for (int kt = 0; kt < 2; ++kt)
#pragma unroll
        for (int r = 0; r < 8; ++r) bm = fmaxf(bm, st[kt][qt][r]);
      bm = fmaxf(bm, __shfl_xor(bm, 16, 32));
      const float mnew = fmaxf(mrow[qt], bm);
      const float alpha = fexp2(mrow[qt] - mnew);
      mrow[qt] = mnew;

      float e0[8], e1[8];
      float sum = 0.f;
#pragma unroll
      for (int r = 0; r < 8; ++r) {
        e0[r] = fexp2(st[0][qt][r] - mnew);
        e1[r] = fexp2(st[1][qt][r] - mnew);
        sum += e0[r] + e1[r];
      }
      sum += __shfl_xor(sum, 16, 32);
      lrow[qt] = lrow[qt] * alpha + sum;

      // Build P^T B-fragment. Half0 lane needs keys 0..15: own tile0 (keys 0-7)
      // + partner's tile0 (keys 8-15). Half1 lane needs keys 16..31: partner's
      // tile1 (16-23) + own tile1 (24-31). Pack to bf16 dwords first, then one
      // xor-16 shuffle per dword.
      FragBF p;
#pragma unroll
      for (int j = 0; j < 4; ++j) {
        const unsigned own0 = pk_bf16_trunc(e0[2 * j], e0[2 * j + 1]);
        const unsigned own1 = pk_bf16_trunc(e1[2 * j], e1[2 * j + 1]);
        const unsigned rem0 = __shfl_xor(own0, 16, 32);
        const unsigned rem1 = __shfl_xor(own1, 16, 32);
        p.d[j]     = hf ? rem1 : own0;
        p.d[4 + j] = hf ? own1 : rem0;
      }
      pf[qt] = p.bf;

      // rescale accumulated O only when the running max actually moved
      if (__any(alpha < 1.0f)) {
#pragma unroll
        for (int dt = 0; dt < 4; ++dt)
#pragma unroll
          for (int r = 0; r < 8; ++r) oacc[dt][qt][r] *= alpha;
      }
    }

    // O^T += V^T(16 dims x 32 keys) @ P^T(32 keys x 16 queries)
#pragma unroll
    for (int dt = 0; dt < 4; ++dt) {
      v16bf vf = load_fragA(Vh, dt * 16 + lr, Sq, kb, lane);
#pragma unroll
      for (int qt = 0; qt < QPT; ++qt)
        oacc[dt][qt] = wmma_bf16(vf, pf[qt], oacc[dt][qt]);
    }
  }

  // epilogue: ctx[b*4096+seq][h*64+d] bf16, row-major [8192][512]
  const int b = bh >> 3, h = bh & 7;
#pragma unroll
  for (int qt = 0; qt < QPT; ++qt) {
    const float inv = 1.0f / lrow[qt];
    const int seq = q0 + 16 * qt + lr;
    const size_t rowoff = ((size_t)(b * Sq + seq)) * Dm + h * DHd + 8 * hf;
#pragma unroll
    for (int dt = 0; dt < 4; ++dt) {
      v4ui o;
#pragma unroll
      for (int j = 0; j < 4; ++j)
        o[j] = pk_bf16_trunc(oacc[dt][qt][2 * j] * inv, oacc[dt][qt][2 * j + 1] * inv);
      *(v4ui*)(ctx + rowoff + dt * 16) = o;
    }
  }
}

// ---------------- host ----------------

extern "C" void kernel_launch(void* const* d_in, const int* in_sizes, int n_in,
                              void* d_out, int out_size, void* d_ws, size_t ws_size,
                              hipStream_t stream) {
  (void)in_sizes; (void)n_in; (void)out_size; (void)ws_size;
  const float* X  = (const float*)d_in[0];
  const float* Wq = (const float*)d_in[1];
  const float* bq = (const float*)d_in[2];
  const float* Wk = (const float*)d_in[3];
  const float* bk = (const float*)d_in[4];
  const float* Wv = (const float*)d_in[5];
  const float* bv = (const float*)d_in[6];
  const float* Wo = (const float*)d_in[7];
  const float* bo = (const float*)d_in[8];

  char* ws = (char*)d_ws;
  size_t off = 0;
  auto alloc = [&](size_t bytes) {
    char* p = ws + off;
    off += (bytes + 255) & ~(size_t)255;
    return p;
  };
  u16* Xbf = (u16*)alloc((size_t)Mr * Dm * sizeof(u16));   // also reused as ctx
  u16* WqT = (u16*)alloc((size_t)Dm * Dm * sizeof(u16));
  u16* WkT = (u16*)alloc((size_t)Dm * Dm * sizeof(u16));
  u16* WvT = (u16*)alloc((size_t)Dm * Dm * sizeof(u16));
  u16* WoT = (u16*)alloc((size_t)Dm * Dm * sizeof(u16));
  u16* Qbf = (u16*)alloc((size_t)Mr * Dm * sizeof(u16));
  u16* Kbf = (u16*)alloc((size_t)Mr * Dm * sizeof(u16));
  u16* Vtb = (u16*)alloc((size_t)Mr * Dm * sizeof(u16));
  u16* Ctx = Xbf;  // X is dead after the V projection; stream order makes this safe

  const int nX = Mr * Dm;
  cvt_f32_to_bf16<<<nX / (256 * 8), 256, 0, stream>>>(X, Xbf);
  transpose_w_bf16<<<Dm * Dm / 256, 256, 0, stream>>>(Wq, WqT);
  transpose_w_bf16<<<Dm * Dm / 256, 256, 0, stream>>>(Wk, WkT);
  transpose_w_bf16<<<Dm * Dm / 256, 256, 0, stream>>>(Wv, WvT);
  transpose_w_bf16<<<Dm * Dm / 256, 256, 0, stream>>>(Wo, WoT);

  dim3 gg(Mr / 128, Dm / 128);
  // Q pre-scale = (1/sqrt(64)) * log2(e): moves softmax into exp2 domain for free
  gemm_bf16<<<gg, 256, 0, stream>>>(Xbf, WqT, bq, Qbf, MODE_QK, 0.18033688011112042f);
  gemm_bf16<<<gg, 256, 0, stream>>>(Xbf, WkT, bk, Kbf, MODE_QK, 1.0f);
  gemm_bf16<<<gg, 256, 0, stream>>>(Xbf, WvT, bv, Vtb, MODE_VT, 1.0f);

  flash_attn<<<dim3(Sq / 128, Bsz * Hn), 128, 0, stream>>>(Qbf, Kbf, Vtb, Ctx);

  gemm_bf16<<<gg, 256, 0, stream>>>(Ctx, WoT, bo, d_out, MODE_F32, 1.0f);
}
